// CorrespondenceGenerationArch_53609781788765
// MI455X (gfx1250) — matrix operand order repaired and checked by reading, not compiled, not run on hardware
//
#include <hip/hip_runtime.h>
#include <hip/hip_bf16.h>
#include <math.h>

typedef _Float16 h16;
typedef __attribute__((ext_vector_type(8)))  _Float16 v8h;
typedef __attribute__((ext_vector_type(16))) _Float16 v16h;
typedef __attribute__((ext_vector_type(8)))  float    v8f;

#define CCH   256
#define HF    80
#define WF    80
#define H2    78
#define NP    6084        /* 78*78 patches */
#define NPAD  6096        /* 381*16        */
#define NTILE 381
#define SAPITCH 2312      /* 2304 + 8 halves pad: row stride 1156 dw == 4 mod 64 banks */

static constexpr size_t alignup(size_t x) { return (x + 255) & ~(size_t)255; }

// ---------------- workspace layout (bytes) ----------------
static constexpr size_t WS_TOPIDX = 0;                                        // int   [2][3][NP]
static constexpr size_t WS_TOPVAL = alignup(WS_TOPIDX + 2ull*3*NP*4);         // float [2][3][NP]
static constexpr size_t WS_PIXSQ  = alignup(WS_TOPVAL + 2ull*3*NP*4);         // float [2][2][6400]
static constexpr size_t WS_INV    = alignup(WS_PIXSQ  + 2ull*2*HF*WF*4);      // float [2][2][NPAD]
static constexpr size_t WS_FNORM  = alignup(WS_INV    + 2ull*2*NPAD*4);       // h16   [2][2][6400*256]
static constexpr size_t WS_IMGN   = alignup(WS_FNORM  + 2ull*2*HF*WF*CCH*2);  // float [2][3][320][320]
static constexpr size_t WS_C1     = alignup(WS_IMGN   + 2ull*3*320*320*4);    // float [2][64][320][320]
static constexpr size_t WS_P1     = alignup(WS_C1     + 2ull*64*320*320*4);   // float [2][64][160][160]
static constexpr size_t WS_C2     = WS_C1;                                    // alias (c1 dead by then)
static constexpr size_t WS_P2     = WS_C1 + 2ull*128*160*160*4;               // alias inside c1 region

// ============================================================
// 1) Per-pixel channel normalization -> HWC f16, plus sum-sq of
//    the normalized vector per pixel (for patch norms).
// ============================================================
__global__ void k_feat_norm(const float* __restrict__ d1, const float* __restrict__ d2,
                            h16* __restrict__ fn, float* __restrict__ pixsq)
{
  int p = blockIdx.x * blockDim.x + threadIdx.x;   // pixel 0..6399
  if (p >= HF * WF) return;
  int m = blockIdx.y;                              // 0 = input feat, 1 = ref feat
  int b = blockIdx.z;
  const float* src = (m == 0 ? d1 : d2) + (size_t)b * CCH * HF * WF;
  h16* dst = fn + (size_t)(b * 2 + m) * (HF * WF * CCH);
  float ss = 0.f;
  for (int c = 0; c < CCH; ++c) { float v = src[c * HF * WF + p]; ss = fmaf(v, v, ss); }
  float inv = 1.f / fmaxf(sqrtf(ss), 1e-12f);
  float os = 0.f;
  for (int c = 0; c < CCH; ++c) {
    float v = src[c * HF * WF + p] * inv;
    os = fmaf(v, v, os);
    dst[(size_t)p * CCH + c] = (h16)v;
  }
  pixsq[(b * 2 + m) * HF * WF + p] = os;
}

// ============================================================
// 2) Per-patch inverse norm: 1 / (sqrt(sum of 9 pixel sumsq) + 1e-5)
// ============================================================
__global__ void k_patch_inv(const float* __restrict__ pixsq, float* __restrict__ invn)
{
  int j = blockIdx.x * blockDim.x + threadIdx.x;
  if (j >= NPAD) return;
  int m = blockIdx.y, b = blockIdx.z;
  const float* ps = pixsq + (b * 2 + m) * HF * WF;
  float r = 0.f;
  if (j < NP) {
    int y = j / H2, x = j % H2;
    float s = 0.f;
    #pragma unroll
    for (int ki = 0; ki < 3; ++ki)
      #pragma unroll
      for (int kj = 0; kj < 3; ++kj)
        s += ps[(y + ki) * WF + (x + kj)];
    r = 1.f / (sqrtf(s) + 1e-5f);
  }
  invn[(b * 2 + m) * NPAD + j] = r;
}

// ============================================================
// 3) WMMA patch-correlation + fused top-3.
//    Block = 256 threads = 8 waves; block owns one 16-row M tile.
//    The 16x2304 A tile is staged ONCE in LDS (shared by all 8
//    waves, nt-invariant) -> no scratch spills, no global A reads
//    in the inner loop. (ki,kj) loops stay rolled so LICM cannot
//    hoist the A ds_loads back out of the nt loop.
// ============================================================
__global__ __launch_bounds__(256) void k_match(
    const h16* __restrict__ fnorm, const float* __restrict__ invn,
    float* __restrict__ topVal, int* __restrict__ topIdx)
{
  __shared__ __align__(32) h16 sA[16 * SAPITCH];   // 73,984 B

  int b = blockIdx.y;
  const h16*   A    = fnorm + (size_t)(b * 2 + 0) * (HF * WF * CCH);
  const h16*   Bm   = fnorm + (size_t)(b * 2 + 1) * (HF * WF * CCH);
  const float* invA = invn + (b * 2 + 0) * NPAD;
  const float* invB = invn + (b * 2 + 1) * NPAD;

  int mtile = blockIdx.x;
  int wave = threadIdx.x >> 5;
  int lane = threadIdx.x & 31;
  int hi = lane >> 4, l16 = lane & 15;

  // ---- stage A tile: 16 rows x 9 blocks x 256 ch, as 32B units ----
  for (int u = threadIdx.x; u < 16 * 9 * 16; u += 256) {
    int row = u / 144;
    int rem = u - row * 144;
    int blk = rem >> 4;
    int cu  = rem & 15;
    int gm = mtile * 16 + row;
    int gmc = gm < NP ? gm : NP - 1;
    int ay = gmc / H2, ax = gmc % H2;
    int ki = blk / 3, kj = blk - 3 * (blk / 3);
    v16h val = *(const v16h*)(A + ((ay + ki) * WF + (ax + kj)) * CCH + cu * 16);
    *(v16h*)(sA + row * SAPITCH + blk * 256 + cu * 16) = val;
  }
  __syncthreads();

  // Row scales for this lane's C rows (C row = r + 8*hi)
  float invAr[8];
  #pragma unroll
  for (int r = 0; r < 8; ++r) invAr[r] = invA[mtile * 16 + 8 * hi + r];

  float tv[8][3]; int ti[8][3];
  #pragma unroll
  for (int r = 0; r < 8; ++r) {
    tv[r][0] = tv[r][1] = tv[r][2] = -1e30f;
    ti[r][0] = ti[r][1] = ti[r][2] = 0;
  }

  // A fragment base in LDS for this lane (row = l16, K chunk starts at 16*hi)
  const h16* aBase = sA + l16 * SAPITCH + 16 * hi;

  for (int nt = wave; nt < NTILE; nt += 8) {
    int gn  = nt * 16 + l16;                 // B column for this lane
    int gnc = gn < NP ? gn : NP - 1;
    int by0 = gnc / H2, bx0 = gnc % H2;

    v8f acc = {0.f, 0.f, 0.f, 0.f, 0.f, 0.f, 0.f, 0.f};
    #pragma unroll 1
    for (int ki = 0; ki < 3; ++ki) {
      #pragma unroll 1
      for (int kj = 0; kj < 3; ++kj) {
        const h16* ap = aBase + (ki * 3 + kj) * 256;
        const h16* bp = Bm + ((by0 + ki) * WF + (bx0 + kj)) * CCH + 16 * hi;
        #pragma unroll 2
        for (int c0 = 0; c0 < CCH; c0 += 32) {
          // A 16x32 f16 fragment: lane holds K = 16*hi + {0..15}, contiguous
          v8h alo = *(const v8h*)(ap + c0);        // ds_load_b128
          v8h ahi = *(const v8h*)(ap + c0 + 8);    // ds_load_b128
          v16h av = __builtin_shufflevector(alo, ahi, 0,1,2,3,4,5,6,7,8,9,10,11,12,13,14,15);
          // B 32x16 f16 fragment: lane holds K = 16*hi + {0..15}, contiguous
          v16h bv = *(const v16h*)(bp + c0);
          acc = __builtin_amdgcn_wmma_f32_16x16x32_f16(false, av, false, bv,
                                                       (short)0, acc, false, false);
        }
      }
    }

    if (gn < NP) {
      float sB = invB[gn];
      #pragma unroll
      for (int r = 0; r < 8; ++r) {
        float v = acc[r] * sB * invAr[r];
        if (v > tv[r][0]) { tv[r][2]=tv[r][1]; ti[r][2]=ti[r][1];
                            tv[r][1]=tv[r][0]; ti[r][1]=ti[r][0];
                            tv[r][0]=v; ti[r][0]=gn; }
        else if (v > tv[r][1]) { tv[r][2]=tv[r][1]; ti[r][2]=ti[r][1];
                                 tv[r][1]=v; ti[r][1]=gn; }
        else if (v > tv[r][2]) { tv[r][2]=v; ti[r][2]=gn; }
      }
    }
  }

  // ---- cross-lane / cross-wave top-3 merge, aliasing sA's LDS ----
  __syncthreads();                                  // everyone done reading sA
  float* sval = (float*)sA;                         // [8][32][24] = 24,576 B
  int*   sidx = (int*)((char*)sA + 8 * 32 * 24 * 4);// next 24,576 B (fits in 73,984)
  #pragma unroll
  for (int r = 0; r < 8; ++r)
    #pragma unroll
    for (int s = 0; s < 3; ++s) {
      sval[(wave * 32 + lane) * 24 + r * 3 + s] = tv[r][s];
      sidx[(wave * 32 + lane) * 24 + r * 3 + s] = ti[r][s];
    }
  __syncthreads();

  if (threadIdx.x < 16) {
    int mrow = threadIdx.x;
    int ghi = mrow >> 3, r = mrow & 7;
    float bv0 = -1e30f, bv1 = -1e30f, bv2 = -1e30f;
    int bi0 = 0, bi1 = 0, bi2 = 0;
    for (int w = 0; w < 8; ++w)
      for (int l = 0; l < 16; ++l) {
        int ln = ghi * 16 + l;
        #pragma unroll
        for (int s = 0; s < 3; ++s) {
          float v = sval[(w * 32 + ln) * 24 + r * 3 + s];
          int  id = sidx[(w * 32 + ln) * 24 + r * 3 + s];
          if (v > bv0) { bv2=bv1; bi2=bi1; bv1=bv0; bi1=bi0; bv0=v; bi0=id; }
          else if (v > bv1) { bv2=bv1; bi2=bi1; bv1=v; bi1=id; }
          else if (v > bv2) { bv2=v; bi2=id; }
        }
      }
    int gout = mtile * 16 + mrow;
    if (gout < NP) {
      float* tV = topVal + (size_t)b * 3 * NP;
      int*   tI = topIdx + (size_t)b * 3 * NP;
      tV[0 * NP + gout] = bv0; tV[1 * NP + gout] = bv1; tV[2 * NP + gout] = bv2;
      tI[0 * NP + gout] = bi0; tI[1 * NP + gout] = bi1; tI[2 * NP + gout] = bi2;
    }
  }
}

// ============================================================
// 4) flow / off(9 shifts) / sim assembly for one pyramid level.
//    s in {1,2,4}, HS = 80*s.
// ============================================================
__global__ void k_flow(const int* __restrict__ topIdx, const float* __restrict__ topVal,
                       float* __restrict__ flow, float* __restrict__ off,
                       float* __restrict__ sim, int s, int HS)
{
  int idx = blockIdx.x * blockDim.x + threadIdx.x;
  int total = 2 * 3 * HS * HS;
  if (idx >= total) return;
  int x = idx % HS; int t1 = idx / HS;
  int y = t1 % HS;  int t2 = t1 / HS;
  int k = t2 % 3;   int b = t2 / 3;
  const int*   tI = topIdx + (size_t)(b * 3 + k) * NP;
  const float* tV = topVal + (size_t)(b * 3 + k) * NP;
  float fs = (float)s;

  int y3 = y / s, x3 = x / s;
  float fx = 0.f, fy = 0.f;
  if (y3 < H2 && x3 < H2) {
    int t = tI[y3 * H2 + x3];
    fx = (float)(t % H2 - x3) * fs;
    fy = (float)(t / H2 - y3) * fs;
  }
  size_t fbase = ((size_t)(b * 3 + k) * HS * HS + (size_t)y * HS + x) * 2;
  flow[fbase] = fx; flow[fbase + 1] = fy;

  #pragma unroll
  for (int i = 0; i < 3; ++i)
    #pragma unroll
    for (int j = 0; j < 3; ++j) {
      int ys = y - s * i, xs = x - s * j;
      float ox = 0.f, oy = 0.f;
      if (ys >= 0 && xs >= 0) {
        int y3s = ys / s, x3s = xs / s;
        if (y3s < H2 && x3s < H2) {
          int t = tI[y3s * H2 + x3s];
          ox = (float)(t % H2 - x3s) * fs;
          oy = (float)(t / H2 - y3s) * fs;
        }
      }
      size_t ob = (((size_t)(b * 3 + k) * 9 + (i * 3 + j)) * HS * HS
                   + (size_t)y * HS + x) * 2;
      off[ob] = ox; off[ob + 1] = oy;
    }

  float sv = 0.f;
  if (y3 >= 1 && y3 <= H2 && x3 >= 1 && x3 <= H2)
    sv = tV[(y3 - 1) * H2 + (x3 - 1)];
  sim[(size_t)(b * 3 + k) * HS * HS + (size_t)y * HS + x] = sv;
}

// ============================================================
// 5) VGG path: ImageNet normalize, 3x3 conv + ReLU, 2x2 maxpool
// ============================================================
__global__ void k_imgnorm(const float* __restrict__ img, float* __restrict__ out)
{
  int idx = blockIdx.x * blockDim.x + threadIdx.x;
  if (idx >= 2 * 3 * 320 * 320) return;
  int c = (idx / (320 * 320)) % 3;
  float m  = (c == 0) ? 0.485f : (c == 1 ? 0.456f : 0.406f);
  float sd = (c == 0) ? 0.229f : (c == 1 ? 0.224f : 0.225f);
  out[idx] = (img[idx] - m) / sd;
}

__global__ void k_conv(const float* __restrict__ in, const float* __restrict__ w,
                       const float* __restrict__ bias, float* __restrict__ out,
                       int Cin, int Cout, int Hh, int Ww)
{
  __shared__ float sw[128 * 9];
  int co = blockIdx.y, b = blockIdx.z;
  for (int t = threadIdx.x; t < Cin * 9; t += blockDim.x)
    sw[t] = w[(size_t)co * Cin * 9 + t];
  __syncthreads();
  int sp = blockIdx.x * blockDim.x + threadIdx.x;
  if (sp >= Hh * Ww) return;
  int y = sp / Ww, x = sp % Ww;
  float acc = bias[co];
  const float* inb = in + (size_t)b * Cin * Hh * Ww;
  for (int ci = 0; ci < Cin; ++ci) {
    const float* ip = inb + (size_t)ci * Hh * Ww;
    const float* wp = sw + ci * 9;
    #pragma unroll
    for (int ki = 0; ki < 3; ++ki) {
      int yy = y + ki - 1;
      if (yy < 0 || yy >= Hh) continue;
      #pragma unroll
      for (int kj = 0; kj < 3; ++kj) {
        int xx = x + kj - 1;
        if (xx < 0 || xx >= Ww) continue;
        acc = fmaf(ip[yy * Ww + xx], wp[ki * 3 + kj], acc);
      }
    }
  }
  out[(((size_t)b * Cout + co) * Hh + y) * Ww + x] = fmaxf(acc, 0.f);
}

__global__ void k_pool(const float* __restrict__ in, float* __restrict__ out,
                       int Cn, int Ho, int Wo)
{
  int idx = blockIdx.x * blockDim.x + threadIdx.x;
  int total = 2 * Cn * Ho * Wo;
  if (idx >= total) return;
  int x = idx % Wo; int y = (idx / Wo) % Ho; int cb = idx / (Wo * Ho);
  const float* ip = in + (size_t)cb * (2 * Ho) * (2 * Wo);
  int Wi = 2 * Wo;
  float a = ip[(2 * y) * Wi + 2 * x];
  float bq = ip[(2 * y) * Wi + 2 * x + 1];
  float c = ip[(2 * y + 1) * Wi + 2 * x];
  float d = ip[(2 * y + 1) * Wi + 2 * x + 1];
  out[idx] = fmaxf(fmaxf(a, bq), fmaxf(c, d));
}

// ============================================================
extern "C" void kernel_launch(void* const* d_in, const int* in_sizes, int n_in,
                              void* d_out, int out_size, void* d_ws, size_t ws_size,
                              hipStream_t stream)
{
  (void)in_sizes; (void)n_in; (void)out_size; (void)ws_size;

  const float* d1  = (const float*)d_in[0];
  const float* d2  = (const float*)d_in[1];
  const float* img = (const float*)d_in[2];
  const float* w11 = (const float*)d_in[3];  const float* b11 = (const float*)d_in[4];
  const float* w12 = (const float*)d_in[5];  const float* b12 = (const float*)d_in[6];
  const float* w21 = (const float*)d_in[7];  const float* b21 = (const float*)d_in[8];
  const float* w22 = (const float*)d_in[9];  const float* b22 = (const float*)d_in[10];
  const float* w31 = (const float*)d_in[11]; const float* b31 = (const float*)d_in[12];

  char* ws = (char*)d_ws;
  int*   topIdx = (int*)(ws + WS_TOPIDX);
  float* topVal = (float*)(ws + WS_TOPVAL);
  float* pixsq  = (float*)(ws + WS_PIXSQ);
  float* invn   = (float*)(ws + WS_INV);
  h16*   fnorm  = (h16*)(ws + WS_FNORM);
  float* imgN   = (float*)(ws + WS_IMGN);
  float* c1     = (float*)(ws + WS_C1);
  float* p1     = (float*)(ws + WS_P1);
  float* c2     = (float*)(ws + WS_C2);
  float* p2     = (float*)(ws + WS_P2);

  float* out = (float*)d_out;
  float* o_flow1 = out + 0;         float* o_flow2 = out + 1228800;  float* o_flow3 = out + 1536000;
  float* o_off1  = out + 1612800;   float* o_off2  = out + 12672000; float* o_off3  = out + 15436800;
  float* o_sim1  = out + 16128000;  float* o_sim2  = out + 16742400; float* o_sim3  = out + 16896000;
  float* r1      = out + 16934400;  float* r2      = out + 30041600; float* r3      = out + 36595200;

  dim3 blk(256);

  // --- feature matching path (the 341-GFLOP part, WMMA) ---
  k_feat_norm<<<dim3((HF * WF + 255) / 256, 2, 2), blk, 0, stream>>>(d1, d2, fnorm, pixsq);
  k_patch_inv<<<dim3((NPAD + 255) / 256, 2, 2), blk, 0, stream>>>(pixsq, invn);
  k_match<<<dim3(NTILE, 2), blk, 0, stream>>>(fnorm, invn, topVal, topIdx);

  // --- flow / offset / similarity pyramids ---
  k_flow<<<(2 * 3 * 80 * 80 + 255) / 256, blk, 0, stream>>>(topIdx, topVal, o_flow3, o_off3, o_sim3, 1, 80);
  k_flow<<<(2 * 3 * 160 * 160 + 255) / 256, blk, 0, stream>>>(topIdx, topVal, o_flow2, o_off2, o_sim2, 2, 160);
  k_flow<<<(2 * 3 * 320 * 320 + 255) / 256, blk, 0, stream>>>(topIdx, topVal, o_flow1, o_off1, o_sim1, 4, 320);

  // --- VGG feature extractor (relu1_1, relu2_1, relu3_1) ---
  k_imgnorm<<<(2 * 3 * 320 * 320 + 255) / 256, blk, 0, stream>>>(img, imgN);
  k_conv<<<dim3(400, 64, 2),  blk, 0, stream>>>(imgN, w11, b11, r1, 3, 64, 320, 320);
  k_conv<<<dim3(400, 64, 2),  blk, 0, stream>>>(r1,   w12, b12, c1, 64, 64, 320, 320);
  k_pool<<<(2 * 64 * 160 * 160 + 255) / 256, blk, 0, stream>>>(c1, p1, 64, 160, 160);
  k_conv<<<dim3(100, 128, 2), blk, 0, stream>>>(p1, w21, b21, r2, 64, 128, 160, 160);
  k_conv<<<dim3(100, 128, 2), blk, 0, stream>>>(r2, w22, b22, c2, 128, 128, 160, 160);
  k_pool<<<(2 * 128 * 80 * 80 + 255) / 256, blk, 0, stream>>>(c2, p2, 128, 80, 80);
  k_conv<<<dim3(25, 256, 2),  blk, 0, stream>>>(p2, w31, b31, r3, 128, 256, 80, 80);
}
